// TriangleAttention_89996744721173
// MI455X (gfx1250) — matrix operand order repaired and checked by reading, not compiled, not run on hardware
//
#include <hip/hip_runtime.h>
#include <hip/hip_bf16.h>

// ---------------------------------------------------------------------------
// Triangle attention ("ending") for MI455X / gfx1250.
// BS=2, S1=S2=256, D=128, C=32, H=4.  ~39 GFLOP, scratch fits in 192MB L2 ->
// WMMA-throughput bound.  bf16 v_wmma_f32_16x16x32_bf16 everywhere, flash
// online softmax in WMMA C/D layout, async-to-LDS K/V staging, pre-converted
// transposed bf16 weights, 32-bit fragment pair loads, padded LDS.
// ---------------------------------------------------------------------------

constexpr int kS1 = 256, kS2 = 256, kD = 128, kC = 32, kH = 4, kB = 2;
constexpr int kM  = kB * kS1 * kS2;                       // 131072 rows
constexpr float kQKScale = 0.42044820762685725f;          // 32^-0.25
constexpr float kLnEps   = 1e-5f;

typedef __attribute__((ext_vector_type(16))) __bf16 v16bf;
typedef __attribute__((ext_vector_type(8)))  float  v8f;

union FragA { v16bf v; unsigned short u[16]; unsigned p[8]; };
union FragC { v8f   v; float f[8]; };

static __device__ __forceinline__ unsigned short f2bf(float x) {
  unsigned u = __float_as_uint(x);
  unsigned r = u + 0x7FFFu + ((u >> 16) & 1u);            // round-to-nearest-even
  return (unsigned short)(r >> 16);
}
static __device__ __forceinline__ float bf2f(unsigned short b) {
  return __uint_as_float(((unsigned)b) << 16);
}
static __device__ __forceinline__ unsigned pack2(float a, float b) {
  return (unsigned)f2bf(a) | ((unsigned)f2bf(b) << 16);
}

// ---- CDNA5 async global->LDS (ASYNCcnt, CDNA5 ISA) --------
typedef __attribute__((address_space(3))) const void lds_cv;
static __device__ __forceinline__ unsigned lds_bo(const void* p) {
  return (unsigned)(unsigned long long)(lds_cv*)p;        // 32-bit LDS byte offset
}
static __device__ __forceinline__ void async_cp16(unsigned lds_byte,
                                                  unsigned long long gaddr) {
  asm volatile("global_load_async_to_lds_b128 %0, %1, off"
               :: "v"(lds_byte), "v"(gaddr) : "memory");
}
static __device__ __forceinline__ void wait_async0() {
  asm volatile("s_wait_asynccnt 0x0" ::: "memory");
}

// -------------------------------- LayerNorm --------------------------------
__global__ void ln_kernel(const float* __restrict__ z, const float* __restrict__ w,
                          const float* __restrict__ bb, unsigned short* __restrict__ zln) {
  int gid  = blockIdx.x * blockDim.x + threadIdx.x;
  int m    = gid >> 5;
  int lane = gid & 31;
  if (m >= kM) return;
  const float4 xv = *(const float4*)(z + (size_t)m * kD + lane * 4);
  float x[4] = {xv.x, xv.y, xv.z, xv.w};
  float s = 0.f, sq = 0.f;
#pragma unroll
  for (int u = 0; u < 4; ++u) { s += x[u]; sq += x[u] * x[u]; }
#pragma unroll
  for (int msk = 16; msk >= 1; msk >>= 1) { s += __shfl_xor(s, msk); sq += __shfl_xor(sq, msk); }
  float mu  = s * (1.f / kD);
  float var = sq * (1.f / kD) - mu * mu;
  float rs  = rsqrtf(var + kLnEps);
  int c = lane * 4;
  float y0 = (x[0] - mu) * rs * w[c + 0] + bb[c + 0];
  float y1 = (x[1] - mu) * rs * w[c + 1] + bb[c + 1];
  float y2 = (x[2] - mu) * rs * w[c + 2] + bb[c + 2];
  float y3 = (x[3] - mu) * rs * w[c + 3] + bb[c + 3];
  uint2 o; o.x = pack2(y0, y1); o.y = pack2(y2, y3);
  *(uint2*)(zln + (size_t)m * kD + c) = o;
}

// -------- Weight pre-conversion: f32 [K,N] -> bf16 transposed [N,K] --------
__global__ void convert_wT_kernel(const float* __restrict__ src,
                                  unsigned short* __restrict__ dst, int N, int K) {
  int idx = blockIdx.x * blockDim.x + threadIdx.x;
  if (idx >= N * K) return;
  int n = idx / K, k = idx % K;
  dst[idx] = f2bf(src[(size_t)k * N + n]);
}

// ------------------------- QKV projection (WMMA) ---------------------------
// z_ln[M,128] @ wqkv -> q,k : [b,h,j,i,c]  v : [b,h,j,c,i]
__global__ void qkv_proj_kernel(const unsigned short* __restrict__ zln,
                                const unsigned short* __restrict__ wqkvT, // [384][128]
                                unsigned short* __restrict__ qb,
                                unsigned short* __restrict__ kb,
                                unsigned short* __restrict__ vb) {
  int lane = threadIdx.x & 31, wid = threadIdx.x >> 5;
  int half = lane >> 4, l16 = lane & 15;
  int mtile = blockIdx.x;
  int ntile = blockIdx.y * 8 + wid;            // 0..23 (wave-uniform)
  int n     = ntile * 16 + l16;
  int mrowA = mtile * 16 + l16;
  FragC acc;
#pragma unroll
  for (int r = 0; r < 8; ++r) acc.f[r] = 0.f;
#pragma unroll
  for (int kb_ = 0; kb_ < 4; ++kb_) {
    FragA A, Bm;
#pragma unroll
    for (int v = 0; v < 8; ++v) {
      int ka = kb_ * 32 + 2 * v + half * 16;                      // A 16x32 layout
      A.p[v]  = *(const unsigned*)(zln + (size_t)mrowA * kD + ka);
      int kk  = kb_ * 32 + 2 * v + half * 16;                     // B 32x16 layout
      Bm.p[v] = *(const unsigned*)(wqkvT + (size_t)n * kD + kk);
    }
    acc.v = __builtin_amdgcn_wmma_f32_16x16x32_bf16(false, A.v, false, Bm.v,
                                                    (short)0, acc.v, false, false);
  }
  // scalarize the section select: 16-col tiles never straddle q/k/v bounds
  int sec = __builtin_amdgcn_readfirstlane(ntile >> 3);  // 0=q, 1=k, 2=v
  int hc = n & 127, h = hc >> 5, c = hc & 31;
  size_t heads[8]; int irow[8];
#pragma unroll
  for (int r = 0; r < 8; ++r) {
    int m = mtile * 16 + r + 8 * half;
    int b = m >> 16, rem = m & 65535, i = rem >> 8, j = rem & 255;
    heads[r] = ((size_t)(b * kH + h)) * kS2 + j;
    irow[r]  = i;
  }
  if (sec == 0) {
#pragma unroll
    for (int r = 0; r < 8; ++r)
      qb[(heads[r] * kS1 + irow[r]) * kC + c] = f2bf(acc.f[r] * kQKScale);
  } else if (sec == 1) {
#pragma unroll
    for (int r = 0; r < 8; ++r)
      kb[(heads[r] * kS1 + irow[r]) * kC + c] = f2bf(acc.f[r] * kQKScale);
  } else {
#pragma unroll
    for (int r = 0; r < 8; ++r)
      vb[(heads[r] * kC + c) * kS1 + irow[r]] = f2bf(acc.f[r]);
  }
}

// ------------------------- Gate projection (WMMA) --------------------------
__global__ void gate_proj_kernel(const unsigned short* __restrict__ zln,
                                 const unsigned short* __restrict__ wgT, // [128][128]
                                 const float* __restrict__ wg_b,
                                 unsigned short* __restrict__ glin) {
  int lane = threadIdx.x & 31, wid = threadIdx.x >> 5;
  int half = lane >> 4, l16 = lane & 15;
  int mtile = blockIdx.x;
  int n     = wid * 16 + l16;
  int mrowA = mtile * 16 + l16;
  FragC acc;
#pragma unroll
  for (int r = 0; r < 8; ++r) acc.f[r] = 0.f;
#pragma unroll
  for (int kb_ = 0; kb_ < 4; ++kb_) {
    FragA A, Bm;
#pragma unroll
    for (int v = 0; v < 8; ++v) {
      int ka = kb_ * 32 + 2 * v + half * 16;
      A.p[v]  = *(const unsigned*)(zln + (size_t)mrowA * kD + ka);
      int kk  = kb_ * 32 + 2 * v + half * 16;
      Bm.p[v] = *(const unsigned*)(wgT + (size_t)n * kD + kk);
    }
    acc.v = __builtin_amdgcn_wmma_f32_16x16x32_bf16(false, A.v, false, Bm.v,
                                                    (short)0, acc.v, false, false);
  }
  float bn = wg_b[n];
#pragma unroll
  for (int r = 0; r < 8; ++r) {
    int m = mtile * 16 + r + 8 * half;
    glin[(size_t)m * kD + n] = f2bf(acc.f[r] + bn);
  }
}

// -------------------- Bias projection (H=4 cols, VALU) ---------------------
__global__ void bias_proj_kernel(const unsigned short* __restrict__ zln,
                                 const float* __restrict__ wbias,
                                 float* __restrict__ biasb) {
  int idx = blockIdx.x * blockDim.x + threadIdx.x;   // kM*kH threads
  int m = idx >> 2, h = idx & 3;
  const unsigned* zr = (const unsigned*)(zln + (size_t)m * kD);
  float s = 0.f;
#pragma unroll 8
  for (int d2 = 0; d2 < 64; ++d2) {
    unsigned pr = zr[d2];
    s += bf2f((unsigned short)(pr & 0xFFFFu)) * wbias[(2 * d2) * kH + h];
    s += bf2f((unsigned short)(pr >> 16))     * wbias[(2 * d2 + 1) * kH + h];
  }
  int b = m >> 16, rem = m & 65535, i = rem >> 8, j = rem & 255;
  biasb[(((size_t)(b * kH + h)) * kS2 + j) * kS1 + i] = s;   // bias[b,h,j,k=i]
}

// ---------------------- Attention core (flash, WMMA) -----------------------
// One block per (b,h,j); 16 waves x 16-row i-tiles.  K/V staged via async-LDS.
constexpr int kKStr = 40;    // Kl row stride (ushorts): 80B -> 20-bank step
constexpr int kVStr = 264;   // Vl row stride: 528B -> 4-bank step
constexpr int kPStr = 40;    // Pst row stride

__global__ void __launch_bounds__(512)
attn_kernel(const unsigned short* __restrict__ qb,
            const unsigned short* __restrict__ kb,
            const unsigned short* __restrict__ vb,
            const float* __restrict__ biasb,
            const unsigned short* __restrict__ glin,
            unsigned short* __restrict__ merged) {
  __shared__ __align__(16) unsigned short Kl[kS1 * kKStr];      // 20 KB
  __shared__ __align__(16) unsigned short Vl[kC * kVStr];       // 16.5 KB
  __shared__ __align__(16) float          Bias[kS1];            //  1 KB
  __shared__ __align__(16) unsigned short Pst[16][16 * kPStr];  // 20 KB

  int bid = blockIdx.x;
  int b = bid >> 10, rem = bid & 1023, h = rem >> 8, j = rem & 255;
  size_t head = ((size_t)(b * kH + h)) * kS2 + j;
  const unsigned short* qg = qb + head * (kS1 * kC);
  const unsigned short* kg = kb + head * (kS1 * kC);
  const unsigned short* vg = vb + head * (kC * kS1);
  const float*          bg = biasb + head * kS1;

  int t = threadIdx.x;
  {
    // K: 256 rows x 64B  (row stride 80B in LDS); one 32B half-row per thread
    int krow = t >> 1, khalf = t & 1;
    unsigned long long ga = (unsigned long long)(uintptr_t)kg + krow * 64u + khalf * 32u;
    unsigned lo = lds_bo(Kl) + krow * (kKStr * 2) + khalf * 32u;
    async_cp16(lo, ga);
    async_cp16(lo + 16u, ga + 16u);
    // V: 32 rows x 512B  (row stride 528B in LDS); 32B chunk per thread
    int vrow = t >> 4, vchunk = t & 15;
    unsigned long long gv = (unsigned long long)(uintptr_t)vg + vrow * 512u + vchunk * 32u;
    unsigned lv = lds_bo(Vl) + vrow * (kVStr * 2) + vchunk * 32u;
    async_cp16(lv, gv);
    async_cp16(lv + 16u, gv + 16u);
  }
  if (t < kS1) Bias[t] = bg[t];

  int lane = t & 31, wid = t >> 5;
  int half = lane >> 4, l16 = lane & 15;

  // Q fragment (A 16x32, k-dim = c = 32) straight from global
  FragA Aq;
#pragma unroll
  for (int v = 0; v < 8; ++v) {
    int ka = 2 * v + half * 16;
    Aq.p[v] = *(const unsigned*)(qg + (size_t)(wid * 16 + l16) * kC + ka);
  }

  float mrow[8], lrow[8];
  FragC o0, o1;
#pragma unroll
  for (int r = 0; r < 8; ++r) { mrow[r] = -1e30f; lrow[r] = 0.f; o0.f[r] = 0.f; o1.f[r] = 0.f; }

  wait_async0();          // ASYNCcnt==0 -> K/V resident in LDS
  __syncthreads();

  for (int kc = 0; kc < 8; ++kc) {             // 32 keys per chunk
    FragC s0, s1;
    {
      FragA Bk;
#pragma unroll
      for (int v = 0; v < 8; ++v) {            // tile 0: cols kc*32 + l16
        int cc = 2 * v + half * 16;
        Bk.p[v] = *(const unsigned*)&Kl[(kc * 32 + l16) * kKStr + cc];
      }
#pragma unroll
      for (int r = 0; r < 8; ++r) s0.f[r] = 0.f;
      s0.v = __builtin_amdgcn_wmma_f32_16x16x32_bf16(false, Aq.v, false, Bk.v,
                                                     (short)0, s0.v, false, false);
#pragma unroll
      for (int v = 0; v < 8; ++v) {            // tile 1: cols kc*32 + 16 + l16
        int cc = 2 * v + half * 16;
        Bk.p[v] = *(const unsigned*)&Kl[(kc * 32 + 16 + l16) * kKStr + cc];
      }
#pragma unroll
      for (int r = 0; r < 8; ++r) s1.f[r] = 0.f;
      s1.v = __builtin_amdgcn_wmma_f32_16x16x32_bf16(false, Aq.v, false, Bk.v,
                                                     (short)0, s1.v, false, false);
    }
    float bias0 = Bias[kc * 32 + l16];
    float bias1 = Bias[kc * 32 + 16 + l16];
#pragma unroll
    for (int r = 0; r < 8; ++r) { s0.f[r] += bias0; s1.f[r] += bias1; }

    // online softmax: rows in VGPR dim, cols across 16-lane halves
#pragma unroll
    for (int r = 0; r < 8; ++r) {
      float tmax = fmaxf(s0.f[r], s1.f[r]);
      tmax = fmaxf(tmax, __shfl_xor(tmax, 1));
      tmax = fmaxf(tmax, __shfl_xor(tmax, 2));
      tmax = fmaxf(tmax, __shfl_xor(tmax, 4));
      tmax = fmaxf(tmax, __shfl_xor(tmax, 8));
      float mn   = fmaxf(mrow[r], tmax);
      float corr = __expf(mrow[r] - mn);
      mrow[r] = mn;
      lrow[r] *= corr; o0.f[r] *= corr; o1.f[r] *= corr;
      float p0 = __expf(s0.f[r] - mn);
      float p1 = __expf(s1.f[r] - mn);
      s0.f[r] = p0; s1.f[r] = p1;
      float ps = p0 + p1;
      ps += __shfl_xor(ps, 1); ps += __shfl_xor(ps, 2);
      ps += __shfl_xor(ps, 4); ps += __shfl_xor(ps, 8);
      lrow[r] += ps;
    }

    // P: D-layout -> LDS -> A-layout (per-wave private staging, DS in-order)
    unsigned short* P = Pst[wid];
#pragma unroll
    for (int r = 0; r < 8; ++r) {
      int prow = r + 8 * half;
      P[prow * kPStr + l16]      = f2bf(s0.f[r]);
      P[prow * kPStr + 16 + l16] = f2bf(s1.f[r]);
    }
    FragA Ap;
#pragma unroll
    for (int v = 0; v < 8; ++v) {
      int ka = 2 * v + half * 16;
      Ap.p[v] = *(const unsigned*)&P[l16 * kPStr + ka];
    }
    FragA Bv;
#pragma unroll
    for (int v = 0; v < 8; ++v) {              // c tile 0 (c = l16)
      int kloc = kc * 32 + 2 * v + half * 16;
      Bv.p[v] = *(const unsigned*)&Vl[l16 * kVStr + kloc];
    }
    o0.v = __builtin_amdgcn_wmma_f32_16x16x32_bf16(false, Ap.v, false, Bv.v,
                                                   (short)0, o0.v, false, false);
#pragma unroll
    for (int v = 0; v < 8; ++v) {              // c tile 1 (c = 16 + l16)
      int kloc = kc * 32 + 2 * v + half * 16;
      Bv.p[v] = *(const unsigned*)&Vl[(16 + l16) * kVStr + kloc];
    }
    o1.v = __builtin_amdgcn_wmma_f32_16x16x32_bf16(false, Ap.v, false, Bv.v,
                                                   (short)0, o1.v, false, false);
  }

  // normalize, gate (sigmoid), scatter into merged [b,i,j, c*H+h] (bf16)
#pragma unroll
  for (int r = 0; r < 8; ++r) {
    int i = wid * 16 + r + 8 * half;
    size_t mo  = ((size_t)b * kS1 + i) * kS2 + j;
    float inv  = 1.f / lrow[r];
    size_t e0  = mo * kD + (size_t)l16 * kH + h;
    size_t e1  = mo * kD + (size_t)(16 + l16) * kH + h;
    float g0 = 1.f / (1.f + __expf(-bf2f(glin[e0])));
    float g1 = 1.f / (1.f + __expf(-bf2f(glin[e1])));
    merged[e0] = f2bf(o0.f[r] * inv * g0);
    merged[e1] = f2bf(o1.f[r] * inv * g1);
  }
}

// --------------------------- Output projection -----------------------------
__global__ void out_proj_kernel(const unsigned short* __restrict__ merged,
                                const unsigned short* __restrict__ woT, // [128][128]
                                const float* __restrict__ wo_b,
                                float* __restrict__ out) {
  int lane = threadIdx.x & 31, wid = threadIdx.x >> 5;
  int half = lane >> 4, l16 = lane & 15;
  int mtile = blockIdx.x;
  int n     = wid * 16 + l16;
  int mrowA = mtile * 16 + l16;
  FragC acc;
#pragma unroll
  for (int r = 0; r < 8; ++r) acc.f[r] = 0.f;
#pragma unroll
  for (int kb_ = 0; kb_ < 4; ++kb_) {
    FragA A, Bm;
#pragma unroll
    for (int v = 0; v < 8; ++v) {
      int ka = kb_ * 32 + 2 * v + half * 16;
      A.p[v]  = *(const unsigned*)(merged + (size_t)mrowA * kD + ka);
      int kk  = kb_ * 32 + 2 * v + half * 16;
      Bm.p[v] = *(const unsigned*)(woT + (size_t)n * kD + kk);
    }
    acc.v = __builtin_amdgcn_wmma_f32_16x16x32_bf16(false, A.v, false, Bm.v,
                                                    (short)0, acc.v, false, false);
  }
  float bn = wo_b[n];
#pragma unroll
  for (int r = 0; r < 8; ++r) {
    int m = mtile * 16 + r + 8 * half;
    out[(size_t)m * kD + n] = acc.f[r] + bn;
  }
}

// ------------------------------- launcher ----------------------------------
extern "C" void kernel_launch(void* const* d_in, const int* in_sizes, int n_in,
                              void* d_out, int out_size, void* d_ws, size_t ws_size,
                              hipStream_t stream) {
  const float* zij   = (const float*)d_in[0];
  const float* ln_w  = (const float*)d_in[1];
  const float* ln_b  = (const float*)d_in[2];
  const float* wqkv  = (const float*)d_in[3];
  const float* wbias = (const float*)d_in[4];
  const float* wg_w  = (const float*)d_in[5];
  const float* wg_b  = (const float*)d_in[6];
  const float* wo_w  = (const float*)d_in[7];
  const float* wo_b  = (const float*)d_in[8];
  float* out = (float*)d_out;

  char* ws = (char*)d_ws;
  size_t off = 0;
  auto alloc = [&](size_t bytes) {
    void* p = ws + off;
    off += (bytes + 255) & ~(size_t)255;
    return p;
  };
  unsigned short* zln   = (unsigned short*)alloc((size_t)kM * kD * 2);   // 32 MB
  unsigned short* qbuf  = (unsigned short*)alloc((size_t)kM * kD * 2);   // 32 MB
  unsigned short* kbuf  = (unsigned short*)alloc((size_t)kM * kD * 2);   // 32 MB
  unsigned short* vbuf  = (unsigned short*)alloc((size_t)kM * kD * 2);   // 32 MB
  float*          biasb = (float*)alloc((size_t)kB * kH * kS2 * kS1 * 4);// 2 MB
  unsigned short* glin  = (unsigned short*)alloc((size_t)kM * kD * 2);   // 32 MB
  unsigned short* wqkvT = (unsigned short*)alloc((size_t)384 * kD * 2);
  unsigned short* wgT   = (unsigned short*)alloc((size_t)kD * kD * 2);
  unsigned short* woT   = (unsigned short*)alloc((size_t)kD * kD * 2);
  unsigned short* merged = zln;   // z_ln dead once projections complete

  convert_wT_kernel<<<(384 * kD + 255) / 256, 256, 0, stream>>>(wqkv, wqkvT, 384, kD);
  convert_wT_kernel<<<(kD * kD + 255) / 256, 256, 0, stream>>>(wg_w, wgT, kD, kD);
  convert_wT_kernel<<<(kD * kD + 255) / 256, 256, 0, stream>>>(wo_w, woT, kD, kD);
  ln_kernel<<<kM / 8, 256, 0, stream>>>(zij, ln_w, ln_b, zln);
  qkv_proj_kernel<<<dim3(kM / 16, 3), 256, 0, stream>>>(zln, wqkvT, qbuf, kbuf, vbuf);
  gate_proj_kernel<<<kM / 16, 256, 0, stream>>>(zln, wgT, wg_b, glin);
  bias_proj_kernel<<<kM * kH / 256, 256, 0, stream>>>(zln, wbias, biasb);
  attn_kernel<<<kB * kH * kS2, 512, 0, stream>>>(qbuf, kbuf, vbuf, biasb, glin, merged);
  out_proj_kernel<<<kM / 16, 256, 0, stream>>>(merged, woT, wo_b, out);
}
